// Evolution_22454089023931
// MI455X (gfx1250) — matrix operand order repaired and checked by reading, not compiled, not run on hardware
//
#include <hip/hip_runtime.h>
#include <hip/hip_bf16.h>

typedef __attribute__((ext_vector_type(16))) _Float16 v16h;
typedef __attribute__((ext_vector_type(8)))  float    v8f;

#define L 128
#define D 64
#define NT 16
#define NB 2048

__device__ __forceinline__ float fast_rcp(float x) { return __builtin_amdgcn_rcpf(x); }

// ---------------- WMMA tile GEMM over LDS-resident f16 operands ----------------
// C[M,N](f32,ldc) = A[M,K](f16,lda) * B[K,N](f16,ldb); tiles round-robined over 8 waves.
// Layouts per CDNA5 ISA 7.12.2:
//   A 16x32 f16 : lane = M (0..15, repeated), kbase = (lane>>4)*8, elem e -> K = kbase + (e&7) + (e>=8?16:0)
//   B 32x16 f16 : lane -> N = lane&15, kbase = (lane>>4)*8, elem e -> K = kbase + (e&7) + (e>=8?16:0)
//   C/D 16x16 f32: col = lane&15, row = vgpr + 8*(lane>>4)
__device__ __forceinline__ void gemm_lds(const _Float16* __restrict__ A, int lda,
                                         const _Float16* __restrict__ B, int ldb,
                                         float* __restrict__ C, int ldc,
                                         int M, int N, int K, int wv, int lane) {
  const int nt = N >> 4;
  const int tiles = (M >> 4) * nt;
  for (int t = wv; t < tiles; t += 8) {
    const int m0 = (t / nt) << 4;
    const int n0 = (t % nt) << 4;
    v8f acc = {};
    for (int k0 = 0; k0 < K; k0 += 32) {
      v16h af{}, bf{};
      const int m  = m0 + (lane & 15);
      const int kb = k0 + ((lane >> 4) << 3);
      const _Float16* Ap = A + m * lda + kb;
#pragma unroll
      for (int e = 0; e < 16; ++e)
        af[e] = Ap[(e & 7) + ((e & 8) << 1)];
      const int kk = k0 + ((lane >> 4) << 3);
      const _Float16* Bp = B + kk * ldb + n0 + (lane & 15);
#pragma unroll
      for (int e = 0; e < 16; ++e) bf[e] = Bp[((e & 7) + ((e & 8) << 1)) * ldb];
      acc = __builtin_amdgcn_wmma_f32_16x16x32_f16(false, af, false, bf,
                                                   (short)0, acc, false, false);
    }
    const int row = m0 + ((lane >> 4) << 3);
    const int col = n0 + (lane & 15);
#pragma unroll
    for (int r = 0; r < 8; ++r) C[(row + r) * ldc + col] = acc[r];
  }
}

// ---------------- argument bundles ----------------
struct PrepArgs {
  const float *text, *W1, *b1, *lng, *lnb, *W2, *b2;
  const float *Wtq, *btq, *Wtk, *btk, *Wtv, *btv;
  const float *Wtr1, *Wtr2, *Wg, *Wf, *Wiq, *Wik, *Wiv, *Wout;
  float *proj, *temb;
  _Float16 *tqh, *tkT, *tvh, *tr1t, *tr2t, *g1t, *f1t, *iqt, *ikt, *ivt, *outt;
};

struct MainArgs {
  const float* contour;
  float* out;
  const float *proj, *temb;
  const _Float16 *tqh, *tkT, *tvh, *tr1t, *tr2t, *g1t, *f1t, *iqt, *ikt, *ivt, *outt;
  const float *Wg, *Wf;
  const float *btr1, *btr2, *bg, *bf, *biq, *bik, *biv, *bout;
  const float *trg, *trb, *fug, *fub, *nmg, *nmb, *l1g, *l1b, *l2g, *l2b;
};

// ---------------- prep: text pipeline + f16 transposed weights ----------------
__global__ void __launch_bounds__(256) prep_kernel(PrepArgs P) {
  __shared__ float st1[NT * 256];
  __shared__ float sp[NT * D];
  const int tid = threadIdx.x;

  if (blockIdx.x == 0) {
    for (int i = tid; i < NT * 256; i += 256) {
      const int n = i >> 8, o = i & 255;
      float s = P.b1[o];
      const float* tr = P.text + n * 768;
      const float* wr = P.W1 + o * 768;
      for (int j = 0; j < 768; ++j) s += tr[j] * wr[j];
      st1[i] = s;
    }
    __syncthreads();
    if (tid < NT) {
      float m = 0.f, ss = 0.f;
      for (int o = 0; o < 256; ++o) { float t = st1[tid * 256 + o]; m += t; ss += t * t; }
      m *= (1.f / 256.f);
      const float rstd = rsqrtf(ss * (1.f / 256.f) - m * m + 1e-5f);
      for (int o = 0; o < 256; ++o) {
        float t = st1[tid * 256 + o];
        st1[tid * 256 + o] = fmaxf((t - m) * rstd * P.lng[o] + P.lnb[o], 0.f);
      }
    }
    __syncthreads();
    for (int i = tid; i < NT * D; i += 256) {
      const int n = i >> 6, o = i & 63;
      float s = P.b2[o];
      for (int j = 0; j < 256; ++j) s += st1[n * 256 + j] * P.W2[o * 256 + j];
      sp[i] = s;
    }
    __syncthreads();
    for (int i = tid; i < NT * D; i += 256) {
      const int n = i >> 6, o = i & 63;
      float q = P.btq[o], k = P.btk[o], v = P.btv[o];
      for (int j = 0; j < D; ++j) {
        const float x = sp[n * D + j];
        q += x * P.Wtq[o * D + j];
        k += x * P.Wtk[o * D + j];
        v += x * P.Wtv[o * D + j];
      }
      P.tqh[i]           = (_Float16)q;
      P.tkT[o * NT + n]  = (_Float16)k;   // [64][16]
      P.tvh[n * D + o]   = (_Float16)v;   // [32][64] rows 0..15
      P.proj[i] = sp[i];
    }
    for (int i = tid; i < NT * D; i += 256) P.tvh[NT * D + i] = (_Float16)0.f; // K-pad rows
    if (tid < NT) {
      float ss = 0.f;
      for (int d = 0; d < D; ++d) ss += sp[tid * D + d] * sp[tid * D + d];
      const float inv = 1.f / fmaxf(sqrtf(ss), 1e-12f);
      for (int d = 0; d < D; ++d) P.temb[tid * D + d] = sp[tid * D + d] * inv;
    }
  }

  // f16 transposed weights (B operands, row-major [K][N])
  const int TOT = 8192 * 3 + 4096 * 5;
  for (int idx = blockIdx.x * 256 + tid; idx < TOT; idx += gridDim.x * 256) {
    if (idx < 8192)       { int t = idx;          int k = t >> 7, n = t & 127; P.tr1t[t] = (_Float16)P.Wtr1[n * 64 + k]; }
    else if (idx < 16384) { int t = idx - 8192;   int k = t >> 6, n = t & 63;  P.tr2t[t] = (_Float16)P.Wtr2[n * 128 + k]; }
    else if (idx < 24576) { int t = idx - 16384;  int k = t >> 6, n = t & 63;  P.outt[t] = (_Float16)P.Wout[n * 128 + k]; }
    else if (idx < 28672) { int t = idx - 24576;  int k = t >> 6, n = t & 63;  P.g1t[t]  = (_Float16)P.Wg[n * 128 + k]; }
    else if (idx < 32768) { int t = idx - 28672;  int k = t >> 6, n = t & 63;  P.f1t[t]  = (_Float16)P.Wf[n * 128 + k]; }
    else if (idx < 36864) { int t = idx - 32768;  int k = t >> 6, n = t & 63;  P.iqt[t]  = (_Float16)P.Wiq[n * 64 + k]; }
    else if (idx < 40960) { int t = idx - 36864;  int k = t >> 6, n = t & 63;  P.ikt[t]  = (_Float16)P.Wik[n * 64 + k]; }
    else                  { int t = idx - 40960;  int k = t >> 6, n = t & 63;  P.ivt[t]  = (_Float16)P.Wiv[n * 64 + k]; }
  }
}

// ---------------- fused main kernel: one block per batch ----------------
constexpr unsigned LDS_MAIN = 298496;

__global__ void __launch_bounds__(256) fused_main(MainArgs A) {
  extern __shared__ __align__(16) char smem[];
  float*    sX   = (float*)(smem + 0);         // [128][64]  contour -> img (residual)
  _Float16* sXh  = (_Float16*)(smem + 32768);  // [128][128] f16 A staging
  _Float16* sW   = (_Float16*)(smem + 65536);  // [128][128] f16 B staging
  float*    sF   = (float*)(smem + 98304);     // [128][128] f32 GEMM out
  float*    sV   = (float*)(smem + 163840);    // [128][64]  f32 scratch
  _Float16* sGh  = (_Float16*)(smem + 196608); // [128][64]  gate f16
  _Float16* sKT  = (_Float16*)(smem + 212992); // [64][128]  ik^T f16
  _Float16* sQh  = (_Float16*)(smem + 229376); // [128][64]  iq f16
  _Float16* sVh  = (_Float16*)(smem + 245760); // [128][64]  iv f16
  float*    slog = (float*)(smem + 262144);    // [128][16]
  _Float16* sAh  = (_Float16*)(smem + 270336); // [128][32]  a_i2t, K-padded
  float*    st2l = (float*)(smem + 278528);    // [16][128]
  _Float16* st2A = (_Float16*)(smem + 286720); // [16][128]
  float*    st2o = (float*)(smem + 290816);    // [16][64]
  _Float16* stq  = (_Float16*)(smem + 294912); // [16][64]
  float*    sred = (float*)(smem + 296960);
  float* mean    = sred;        float* matched = sred + 64;
  float* gvec    = sred + 128;  float* fvec    = sred + 192;
  float* ssim    = sred + 256;  float* snrm    = sred + 272;
  int*   sidx    = (int*)(sred + 273);

  const int tid = threadIdx.x, lane = tid & 31, wv = tid >> 5;
  const int b = blockIdx.x;
  const float* Xg = A.contour + (size_t)b * (L * D);
  const float scale = 0.125f;

  // S1: load contour tile (f32 + f16 copy)
  for (int i = tid * 4; i < L * D; i += 256 * 4) {
    const float4 v = *(const float4*)(Xg + i);
    *(float4*)(sX + i) = v;
    const int r = i >> 6, c = i & 63;
    sXh[r * 128 + c + 0] = (_Float16)v.x; sXh[r * 128 + c + 1] = (_Float16)v.y;
    sXh[r * 128 + c + 2] = (_Float16)v.z; sXh[r * 128 + c + 3] = (_Float16)v.w;
  }
  __syncthreads();

  // S2: matched text vector (mean -> normalize -> argmax sim)
  if (tid < D) { float s = 0.f; for (int r = 0; r < L; ++r) s += sX[r * D + tid]; mean[tid] = s * (1.f / L); }
  __syncthreads();
  if (tid == 0) { float ss = 0.f; for (int d = 0; d < D; ++d) ss += mean[d] * mean[d]; snrm[0] = fmaxf(sqrtf(ss), 1e-12f); }
  __syncthreads();
  if (tid < NT) {
    const float inv = fast_rcp(snrm[0]); float s = 0.f;
    for (int d = 0; d < D; ++d) s += mean[d] * inv * A.temb[tid * D + d];
    ssim[tid] = s;
  }
  __syncthreads();
  if (tid == 0) { int bi = 0; float bv = ssim[0]; for (int n = 1; n < NT; ++n) if (ssim[n] > bv) { bv = ssim[n]; bi = n; } sidx[0] = bi; }
  __syncthreads();
  const int midx = sidx[0];
  if (tid < D) matched[tid] = A.proj[midx * D + tid];
  __syncthreads();
  if (tid < D) {  // attended == matched -> fold second halves of gate/fus weights
    float g = 0.f, f = 0.f;
    for (int j = 0; j < D; ++j) { const float mj = matched[j]; g += mj * A.Wg[tid * 128 + 64 + j]; f += mj * A.Wf[tid * 128 + 64 + j]; }
    gvec[tid] = g; fvec[tid] = f;
  }
  for (int i = tid; i < 64 * 64; i += 256) sW[i] = A.g1t[i];
  __syncthreads();

  // S3: gate = sigmoid(X*Wg1^T + gvec + b)
  gemm_lds(sXh, 128, sW, 64, sV, 64, 128, 64, 64, wv, lane);
  __syncthreads();
  for (int i = tid; i < L * D; i += 256) {
    const int c = i & 63;
    const float z = sV[i] + A.bg[c] + gvec[c];
    sGh[i] = (_Float16)fast_rcp(1.f + __expf(-z));   // v_rcp_f32, no precise-div sequence
  }
  for (int i = tid; i < 64 * 128; i += 256) sW[i] = A.tr1t[i];
  __syncthreads();

  // S4: tr1 (64 -> 128)
  gemm_lds(sXh, 128, sW, 128, sF, 128, 128, 128, 64, wv, lane);
  __syncthreads();

  // S5: LN(128) + ReLU -> sXh (full width)
  if (tid < L) {
    const int r = tid; float m = 0.f, s = 0.f;
    for (int c = 0; c < 128; ++c) { const float t = sF[r * 128 + c] + A.btr1[c]; m += t; s += t * t; }
    m *= (1.f / 128.f);
    const float rstd = rsqrtf(s * (1.f / 128.f) - m * m + 1e-5f);
    for (int c = 0; c < 128; ++c) {
      const float t = sF[r * 128 + c] + A.btr1[c];
      sXh[r * 128 + c] = (_Float16)fmaxf((t - m) * rstd * A.trg[c] + A.trb[c], 0.f);
    }
  }
  for (int i = tid; i < 128 * 64; i += 256) sW[i] = A.tr2t[i];
  __syncthreads();

  // S6: tr2 (128 -> 64)
  gemm_lds(sXh, 128, sW, 64, sV, 64, 128, 64, 128, wv, lane);
  __syncthreads();

  // S7: tr (with bias) -> f16 A operand
  for (int i = tid; i < L * D; i += 256) {
    const int r = i >> 6, c = i & 63;
    sXh[r * 128 + c] = (_Float16)(sV[i] + A.btr2[c]);
  }
  for (int i = tid; i < 64 * 64; i += 256) sW[i] = A.f1t[i];
  __syncthreads();

  // S8: fus1, then fused/gate/img (two LayerNorms) -> img in sX (f32) and sXh (f16)
  gemm_lds(sXh, 128, sW, 64, sF, 64, 128, 64, 64, wv, lane);
  __syncthreads();
  if (tid < L) {
    const int r = tid; float m = 0.f, s = 0.f;
    for (int c = 0; c < D; ++c) { const float t = sF[r * 64 + c] + A.bf[c] + fvec[c]; m += t; s += t * t; }
    m *= (1.f / 64.f);
    const float rstd = rsqrtf(s * (1.f / 64.f) - m * m + 1e-5f);
    float m2 = 0.f, s2 = 0.f;
    for (int c = 0; c < D; ++c) {
      const float t = sF[r * 64 + c] + A.bf[c] + fvec[c];
      const float fz = fmaxf((t - m) * rstd * A.fug[c] + A.fub[c], 0.f);
      const float x = sX[r * 64 + c] + (float)sGh[r * 64 + c] * fz;
      sV[r * 64 + c] = x; m2 += x; s2 += x * x;
    }
    m2 *= (1.f / 64.f);
    const float rstd2 = rsqrtf(s2 * (1.f / 64.f) - m2 * m2 + 1e-5f);
    for (int c = 0; c < D; ++c) {
      const float img = (sV[r * 64 + c] - m2) * rstd2 * A.nmg[c] + A.nmb[c];
      sX[r * 64 + c] = img;
      sXh[r * 128 + c] = (_Float16)img;
    }
  }
  for (int i = tid; i < 64 * 64; i += 256) sW[i] = A.iqt[i];
  __syncthreads();

  // S9: iq
  gemm_lds(sXh, 128, sW, 64, sF, 64, 128, 64, 64, wv, lane);
  __syncthreads();
  for (int i = tid; i < L * D; i += 256) { const int c = i & 63; sQh[i] = (_Float16)(sF[i] + A.biq[c]); }
  for (int i = tid; i < 64 * 64; i += 256) sW[i] = A.ikt[i];
  __syncthreads();

  // S10: ik -> transposed f16
  gemm_lds(sXh, 128, sW, 64, sF, 64, 128, 64, 64, wv, lane);
  __syncthreads();
  for (int i = tid; i < L * D; i += 256) {
    const int r = i >> 6, c = i & 63;
    sKT[c * 128 + r] = (_Float16)(sF[i] + A.bik[c]);
  }
  for (int i = tid; i < 64 * 64; i += 256) sW[i] = A.ivt[i];
  __syncthreads();

  // S11: iv
  gemm_lds(sXh, 128, sW, 64, sF, 64, 128, 64, 64, wv, lane);
  __syncthreads();
  for (int i = tid; i < L * D; i += 256) { const int c = i & 63; sVh[i] = (_Float16)(sF[i] + A.biv[c]); }
  for (int i = tid; i < 64 * NT; i += 256) sW[i] = A.tkT[i];
  for (int i = tid; i < NT * D; i += 256) stq[i] = A.tqh[i];
  __syncthreads();

  // S12: i2t logits + softmax over 16
  gemm_lds(sQh, 64, sW, NT, slog, NT, 128, NT, 64, wv, lane);
  __syncthreads();
  if (tid < L) {
    const int r = tid; float mx = -1e30f;
    for (int n = 0; n < NT; ++n) mx = fmaxf(mx, slog[r * NT + n] * scale);
    float sum = 0.f;
    for (int n = 0; n < NT; ++n) sum += __expf(slog[r * NT + n] * scale - mx);
    const float inv = fast_rcp(sum);
    for (int n = 0; n < NT; ++n) sAh[r * 32 + n] = (_Float16)(__expf(slog[r * NT + n] * scale - mx) * inv);
    for (int n = NT; n < 32; ++n) sAh[r * 32 + n] = (_Float16)0.f;
  }
  for (int i = tid; i < 32 * 64; i += 256) sW[i] = A.tvh[i];
  __syncthreads();

  // S13: i2t = a @ tv -> concat cols [0,64)
  gemm_lds(sAh, 32, sW, 64, sF, 64, 128, 64, 32, wv, lane);
  __syncthreads();
  for (int i = tid; i < L * D; i += 256) { const int r = i >> 6, c = i & 63; sXh[r * 128 + c] = (_Float16)sF[i]; }
  __syncthreads();

  // S14: t2i logits (tq @ ik^T) + softmax over 128
  gemm_lds(stq, 64, sKT, 128, st2l, 128, NT, 128, 64, wv, lane);
  __syncthreads();
  if (tid < NT) {
    const int r = tid; float mx = -1e30f;
    for (int l = 0; l < L; ++l) mx = fmaxf(mx, st2l[r * 128 + l] * scale);
    float sum = 0.f;
    for (int l = 0; l < L; ++l) sum += __expf(st2l[r * 128 + l] * scale - mx);
    const float inv = fast_rcp(sum);
    for (int l = 0; l < L; ++l) st2A[r * 128 + l] = (_Float16)(__expf(st2l[r * 128 + l] * scale - mx) * inv);
  }
  __syncthreads();

  // S15: t2i = a @ iv ; adaptive pool == 8x row replication -> concat cols [64,128)
  gemm_lds(st2A, 128, sVh, 64, st2o, 64, NT, 64, 128, wv, lane);
  __syncthreads();
  for (int i = tid; i < L * D; i += 256) {
    const int r = i >> 6, c = i & 63;
    sXh[r * 128 + 64 + c] = (_Float16)st2o[(r >> 3) * 64 + c];
  }
  for (int i = tid; i < 128 * 64; i += 256) sW[i] = A.outt[i];
  __syncthreads();

  // S16: ca_out (128 -> 64)
  gemm_lds(sXh, 128, sW, 64, sF, 64, 128, 64, 128, wv, lane);
  __syncthreads();

  // S17: out = LN2(LN1(img + fused2))
  if (tid < L) {
    const int r = tid; float m = 0.f, s = 0.f;
    for (int c = 0; c < D; ++c) { const float x = sX[r * 64 + c] + sF[r * 64 + c] + A.bout[c]; m += x; s += x * x; }
    m *= (1.f / 64.f);
    const float r1 = rsqrtf(s * (1.f / 64.f) - m * m + 1e-5f);
    float m2 = 0.f, s2 = 0.f;
    for (int c = 0; c < D; ++c) {
      const float x = sX[r * 64 + c] + sF[r * 64 + c] + A.bout[c];
      const float y = (x - m) * r1 * A.l1g[c] + A.l1b[c];
      m2 += y; s2 += y * y;
    }
    m2 *= (1.f / 64.f);
    const float r2 = rsqrtf(s2 * (1.f / 64.f) - m2 * m2 + 1e-5f);
    float* outp = A.out + (size_t)b * (L * D) + r * 64;
    for (int c = 0; c < D; ++c) {
      const float x = sX[r * 64 + c] + sF[r * 64 + c] + A.bout[c];
      const float y = (x - m) * r1 * A.l1g[c] + A.l1b[c];
      outp[c] = (y - m2) * r2 * A.l2g[c] + A.l2b[c];
    }
  }
}

// ---------------- launch ----------------
extern "C" void kernel_launch(void* const* d_in, const int* in_sizes, int n_in,
                              void* d_out, int out_size, void* d_ws, size_t ws_size,
                              hipStream_t stream) {
  auto f = [&](int i) { return (const float*)d_in[i]; };
  char* ws = (char*)d_ws;

  float*    proj = (float*)(ws + 0);         // 16*64 f32
  float*    temb = (float*)(ws + 4096);      // 16*64 f32
  _Float16* tqh  = (_Float16*)(ws + 8192);   // 16*64
  _Float16* tkT  = (_Float16*)(ws + 10240);  // 64*16
  _Float16* tvh  = (_Float16*)(ws + 12288);  // 32*64 (K-padded)
  _Float16* tr1t = (_Float16*)(ws + 16384);  // 64*128
  _Float16* tr2t = (_Float16*)(ws + 49152);  // 128*64
  _Float16* g1t  = (_Float16*)(ws + 81920);  // 64*64
  _Float16* f1t  = (_Float16*)(ws + 90112);
  _Float16* iqt  = (_Float16*)(ws + 98304);
  _Float16* ikt  = (_Float16*)(ws + 106496);
  _Float16* ivt  = (_Float16*)(ws + 114688);
  _Float16* outt = (_Float16*)(ws + 122880); // 128*64

  PrepArgs P;
  P.text = f(1);  P.W1 = f(2);  P.b1 = f(3);  P.lng = f(28); P.lnb = f(29);
  P.W2 = f(4);    P.b2 = f(5);
  P.Wtq = f(20);  P.btq = f(21); P.Wtk = f(22); P.btk = f(23); P.Wtv = f(24); P.btv = f(25);
  P.Wtr1 = f(6);  P.Wtr2 = f(8); P.Wg = f(10);  P.Wf = f(12);
  P.Wiq = f(14);  P.Wik = f(16); P.Wiv = f(18); P.Wout = f(26);
  P.proj = proj;  P.temb = temb;
  P.tqh = tqh; P.tkT = tkT; P.tvh = tvh;
  P.tr1t = tr1t; P.tr2t = tr2t; P.g1t = g1t; P.f1t = f1t;
  P.iqt = iqt; P.ikt = ikt; P.ivt = ivt; P.outt = outt;

  MainArgs M;
  M.contour = f(0); M.out = (float*)d_out;
  M.proj = proj; M.temb = temb;
  M.tqh = tqh; M.tkT = tkT; M.tvh = tvh;
  M.tr1t = tr1t; M.tr2t = tr2t; M.g1t = g1t; M.f1t = f1t;
  M.iqt = iqt; M.ikt = ikt; M.ivt = ivt; M.outt = outt;
  M.Wg = f(10); M.Wf = f(12);
  M.btr1 = f(7); M.btr2 = f(9); M.bg = f(11); M.bf = f(13);
  M.biq = f(15); M.bik = f(17); M.biv = f(19); M.bout = f(27);
  M.trg = f(30); M.trb = f(31); M.fug = f(32); M.fub = f(33);
  M.nmg = f(34); M.nmb = f(35); M.l1g = f(36); M.l1b = f(37);
  M.l2g = f(38); M.l2b = f(39);

  prep_kernel<<<dim3(8), dim3(256), 0, stream>>>(P);
  fused_main<<<dim3(NB), dim3(256), LDS_MAIN, stream>>>(M);
}
